// ConditionalSplineFlow_85547158602306
// MI455X (gfx1250) — matrix operand order repaired and checked by reading, not compiled, not run on hardware
//
#include <hip/hip_runtime.h>
#include <hip/hip_bf16.h>

// ---------------- problem constants ----------------
#define B_SZ   32768
#define ZD     64
#define XD     256
#define HID    512
#define TL     8
#define MULT   23
#define NP     (ZD * MULT)      // 1472
#define BM     32               // batch rows per workgroup
#define KA     (ZD + XD)        // 320, fused [cur|x] K
#define THREADS 256
#define TAILC  25.0f

typedef __attribute__((ext_vector_type(16))) __bf16 v16bf;
typedef __attribute__((ext_vector_type(8)))  float  v8f;

#define WMMA_BF16(a, b, c) \
  __builtin_amdgcn_wmma_f32_16x16x32_bf16(false, (a), false, (b), (short)0, (c), false, false)

__device__ __forceinline__ unsigned short f2bf(float f) {
  unsigned int u = __float_as_uint(f);
  u += 0x7FFFu + ((u >> 16) & 1u);          // round to nearest even
  return (unsigned short)(u >> 16);
}

__device__ __forceinline__ v8f zero8() {
  v8f v;
#pragma unroll
  for (int i = 0; i < 8; ++i) v[i] = 0.f;
  return v;
}

union FragU  { v16bf v; unsigned int u[8]; };
union FragQ  { v16bf v; uint4 q[2]; };

// A-matrix 16x32 bf16 fragment from LDS (ISA 7.12.2: lanes 0-15 K=0..7,16..23; lanes 16-31 K=8..15,24..31)
__device__ __forceinline__ v16bf load_a_frag(const unsigned short* sA, int stride,
                                             int r0, int k0, int lane) {
  const int m  = lane & 15;
  const int hh = lane >> 4;
  const unsigned short* p = sA + (r0 + m) * stride + k0 + hh * 8;
  FragU f;
#pragma unroll
  for (int i = 0; i < 8; ++i) {
    const int off = (i < 4) ? (2 * i) : (2 * i + 8);
    f.u[i] = *(const unsigned int*)(p + off);
  }
  return f.v;
}

// B-matrix 32x16 bf16 fragment from global weights stored [N][K] row-major.
// lane n: column n; lanes 0-15 hold K = k0..k0+7,k0+16..k0+23; lanes 16-31 hold K = k0+8..k0+15,k0+24..k0+31.
__device__ __forceinline__ v16bf load_b_frag(const unsigned short* __restrict__ W, int ldw,
                                             int n0, int k0, int lane) {
  const int n  = lane & 15;
  const int hh = lane >> 4;
  const unsigned short* p = W + (size_t)(n0 + n) * ldw + k0 + hh * 8;
  FragQ f;
  f.q[0] = *(const uint4*)(p);
  f.q[1] = *(const uint4*)(p + 16);
  return f.v;
}

// ---------------- prep kernels: mask + convert weights to bf16, stage x, init state ----------------
__global__ void prep_w1(const float* __restrict__ W, unsigned short* __restrict__ O) {
  size_t i = (size_t)blockIdx.x * blockDim.x + threadIdx.x;
  if (i >= (size_t)TL * HID * ZD) return;
  int k = (int)(i % ZD);
  int n = (int)((i / ZD) % HID);
  float m = ((n % 63) >= k) ? 1.f : 0.f;         // (n%63+1) >= (k+1)
  O[i] = f2bf(W[i] * m);
}
__global__ void prep_wc(const float* __restrict__ W, unsigned short* __restrict__ O) {
  size_t i = (size_t)blockIdx.x * blockDim.x + threadIdx.x;
  if (i >= (size_t)TL * HID * XD) return;
  O[i] = f2bf(W[i]);
}
__global__ void prep_w2(const float* __restrict__ W, unsigned short* __restrict__ O) {
  size_t i = (size_t)blockIdx.x * blockDim.x + threadIdx.x;
  if (i >= (size_t)TL * HID * HID) return;
  int k = (int)(i % HID);
  int n = (int)((i / HID) % HID);
  float m = ((n % 63) >= (k % 63)) ? 1.f : 0.f;
  O[i] = f2bf(W[i] * m);
}
__global__ void prep_w3(const float* __restrict__ W, unsigned short* __restrict__ O) {
  size_t i = (size_t)blockIdx.x * blockDim.x + threadIdx.x;
  if (i >= (size_t)TL * NP * HID) return;
  int k = (int)(i % HID);
  int n = (int)((i / HID) % NP);
  float m = ((n / MULT) > (k % 63)) ? 1.f : 0.f; // (n/23+1) > (k%63+1)
  O[i] = f2bf(W[i] * m);
}
__global__ void prep_x(const float* __restrict__ X, unsigned short* __restrict__ O) {
  size_t i = (size_t)blockIdx.x * blockDim.x + threadIdx.x;
  if (i >= (size_t)B_SZ * XD) return;
  O[i] = f2bf(X[i]);
}
__global__ void prep_init(const float* __restrict__ z, float* __restrict__ cur,
                          float* __restrict__ lad) {
  size_t i = (size_t)blockIdx.x * blockDim.x + threadIdx.x;
  if (i < (size_t)B_SZ * ZD) cur[i] = z[i];
  if (i < (size_t)B_SZ)      lad[i] = 0.f;
}

// ---------------- fused layer kernel: 3 WMMA GEMMs + RQ spline ----------------
__global__ __launch_bounds__(THREADS) void layer_kernel(
    const unsigned short* __restrict__ xbf,
    float* __restrict__ cur,
    float* __restrict__ lad,
    const unsigned short* __restrict__ w1m,
    const unsigned short* __restrict__ wcm,
    const unsigned short* __restrict__ w2m,
    const unsigned short* __restrict__ w3m,
    const float* __restrict__ b1, const float* __restrict__ bc,
    const float* __restrict__ b2, const float* __restrict__ b3) {
  extern __shared__ char smem[];
  unsigned short* sA  = (unsigned short*)smem;              // [BM][KA]  bf16  (20.0 KB)
  unsigned short* sH1 = sA + BM * KA;                       // [BM][HID] bf16  (32 KB)
  unsigned short* sH2 = sH1 + BM * HID;                     // [BM][HID] bf16  (32 KB)
  float* sP   = (float*)(sH2 + BM * HID);                   // [BM][NP]  f32   (184 KB)
  float* sZ   = sP + BM * NP;                               // [BM][ZD]  f32   (8 KB)
  float* sLad = sZ + BM * ZD;                               // [BM]      f32

  const int tid  = threadIdx.x;
  const int lane = tid & 31;
  const int wave = tid >> 5;
  const int rowBase = blockIdx.x * BM;

  // ---- stage A panel: [ bf16(cur) | bf16(x) ] and f32 z snapshot ----
  for (int i = tid; i < BM * ZD; i += THREADS) {
    int r = i >> 6, d = i & (ZD - 1);
    float v = cur[(size_t)(rowBase + r) * ZD + d];
    sZ[i] = v;
    sA[r * KA + d] = f2bf(v);
  }
  for (int i = tid; i < BM * XD; i += THREADS) {
    int r = i >> 8, c = i & (XD - 1);
    sA[r * KA + ZD + c] = xbf[(size_t)(rowBase + r) * XD + c];
  }
  if (tid < BM) sLad[tid] = 0.f;
  __syncthreads();

  // ---- GEMM1: h1 = [cur|x] @ [W1m|Wc]^T + b1 + bc   (K = 320), software pipelined ----
  {
    const int n0w = wave * 64;
    v8f acc[2][4];
#pragma unroll
    for (int rt = 0; rt < 2; ++rt)
#pragma unroll
      for (int j = 0; j < 4; ++j) acc[rt][j] = zero8();

    v16bf bcur[4], a0c, a1c;
#pragma unroll
    for (int j = 0; j < 4; ++j) bcur[j] = load_b_frag(w1m, ZD, n0w + j * 16, 0, lane);
    a0c = load_a_frag(sA, KA, 0,  0, lane);
    a1c = load_a_frag(sA, KA, 16, 0, lane);

    const int NSTEP = KA / 32;  // 10
#pragma unroll
    for (int s = 0; s < NSTEP - 1; ++s) {
      const int k1 = (s + 1) * 32;
      const unsigned short* Wsrc = (k1 < ZD) ? w1m : wcm;
      const int ldw = (k1 < ZD) ? ZD : XD;
      const int kk  = (k1 < ZD) ? k1 : (k1 - ZD);
      v16bf bnxt[4];
#pragma unroll
      for (int j = 0; j < 4; ++j) bnxt[j] = load_b_frag(Wsrc, ldw, n0w + j * 16, kk, lane);
      v16bf a0n = load_a_frag(sA, KA, 0,  k1, lane);
      v16bf a1n = load_a_frag(sA, KA, 16, k1, lane);
#pragma unroll
      for (int j = 0; j < 4; ++j) {
        acc[0][j] = WMMA_BF16(a0c, bcur[j], acc[0][j]);
        acc[1][j] = WMMA_BF16(a1c, bcur[j], acc[1][j]);
      }
#pragma unroll
      for (int j = 0; j < 4; ++j) bcur[j] = bnxt[j];
      a0c = a0n; a1c = a1n;
    }
#pragma unroll
    for (int j = 0; j < 4; ++j) {
      acc[0][j] = WMMA_BF16(a0c, bcur[j], acc[0][j]);
      acc[1][j] = WMMA_BF16(a1c, bcur[j], acc[1][j]);
    }

    const int n = lane & 15, hh = lane >> 4;
#pragma unroll
    for (int j = 0; j < 4; ++j) {
      const int col = n0w + j * 16 + n;
      const float bias = b1[col] + bc[col];
#pragma unroll
      for (int rt = 0; rt < 2; ++rt)
#pragma unroll
        for (int r = 0; r < 8; ++r)
          sH1[(rt * 16 + r + hh * 8) * HID + col] = f2bf(acc[rt][j][r] + bias);
    }
  }
  __syncthreads();

  // ---- GEMM2: h2 = relu(h1 @ W2m^T + b2)   (K = 512), software pipelined ----
  {
    const int n0w = wave * 64;
    v8f acc[2][4];
#pragma unroll
    for (int rt = 0; rt < 2; ++rt)
#pragma unroll
      for (int j = 0; j < 4; ++j) acc[rt][j] = zero8();

    v16bf bcur[4], a0c, a1c;
#pragma unroll
    for (int j = 0; j < 4; ++j) bcur[j] = load_b_frag(w2m, HID, n0w + j * 16, 0, lane);
    a0c = load_a_frag(sH1, HID, 0,  0, lane);
    a1c = load_a_frag(sH1, HID, 16, 0, lane);

    const int NSTEP = HID / 32; // 16
#pragma unroll
    for (int s = 0; s < NSTEP - 1; ++s) {
      const int k1 = (s + 1) * 32;
      v16bf bnxt[4];
#pragma unroll
      for (int j = 0; j < 4; ++j) bnxt[j] = load_b_frag(w2m, HID, n0w + j * 16, k1, lane);
      v16bf a0n = load_a_frag(sH1, HID, 0,  k1, lane);
      v16bf a1n = load_a_frag(sH1, HID, 16, k1, lane);
#pragma unroll
      for (int j = 0; j < 4; ++j) {
        acc[0][j] = WMMA_BF16(a0c, bcur[j], acc[0][j]);
        acc[1][j] = WMMA_BF16(a1c, bcur[j], acc[1][j]);
      }
#pragma unroll
      for (int j = 0; j < 4; ++j) bcur[j] = bnxt[j];
      a0c = a0n; a1c = a1n;
    }
#pragma unroll
    for (int j = 0; j < 4; ++j) {
      acc[0][j] = WMMA_BF16(a0c, bcur[j], acc[0][j]);
      acc[1][j] = WMMA_BF16(a1c, bcur[j], acc[1][j]);
    }

    const int n = lane & 15, hh = lane >> 4;
#pragma unroll
    for (int j = 0; j < 4; ++j) {
      const int col = n0w + j * 16 + n;
      const float bias = b2[col];
#pragma unroll
      for (int rt = 0; rt < 2; ++rt)
#pragma unroll
        for (int r = 0; r < 8; ++r)
          sH2[(rt * 16 + r + hh * 8) * HID + col] = f2bf(fmaxf(acc[rt][j][r] + bias, 0.f));
    }
  }
  __syncthreads();

  // ---- GEMM3: p = h2 @ W3m^T + b3  (N = 1472, 184 tiles over 8 waves), pipelined ----
  for (int tt = wave; tt < 2 * (NP / 16); tt += 8) {
    const int rt = tt & 1;
    const int n0 = (tt >> 1) * 16;
    v8f acc = zero8();
    v16bf bcur = load_b_frag(w3m, HID, n0, 0, lane);
    v16bf acur = load_a_frag(sH2, HID, rt * 16, 0, lane);
#pragma unroll
    for (int s = 0; s < (HID / 32) - 1; ++s) {
      const int k1 = (s + 1) * 32;
      v16bf bn = load_b_frag(w3m, HID, n0, k1, lane);
      v16bf an = load_a_frag(sH2, HID, rt * 16, k1, lane);
      acc = WMMA_BF16(acur, bcur, acc);
      acur = an; bcur = bn;
    }
    acc = WMMA_BF16(acur, bcur, acc);

    const int n = lane & 15, hh = lane >> 4;
    const int col = n0 + n;
    const float bias = b3[col];
#pragma unroll
    for (int r = 0; r < 8; ++r)
      sP[(rt * 16 + r + hh * 8) * NP + col] = acc[r] + bias;
  }
  __syncthreads();

  // ---- RQ spline (forward) per (row, dim) ----
  const float INVSH = 0.04419417382415922f;     // 1/sqrt(512)
  for (int pidx = tid; pidx < BM * ZD; pidx += THREADS) {
    const int r = pidx >> 6, d = pidx & (ZD - 1);
    const float* pp = sP + r * NP + d * MULT;
    const float zv = sZ[r * ZD + d];

    float e[8], cw[9], ch[9], der[9];
    // widths softmax -> cumulative bin edges
    float mx = -1e30f, s = 0.f;
#pragma unroll
    for (int i = 0; i < 8; ++i) mx = fmaxf(mx, pp[i] * INVSH);
#pragma unroll
    for (int i = 0; i < 8; ++i) { e[i] = expf(pp[i] * INVSH - mx); s += e[i]; }
    float cum = 0.f, invs = 1.f / s;
    cw[0] = -TAILC;
#pragma unroll
    for (int i = 0; i < 8; ++i) {
      cum += 0.001f + 0.992f * e[i] * invs;     // MIN_W + (1-8*MIN_W)*softmax
      cw[i + 1] = -TAILC + 2.f * TAILC * cum;
    }
    cw[8] = TAILC;
    // heights
    mx = -1e30f; s = 0.f;
#pragma unroll
    for (int i = 0; i < 8; ++i) mx = fmaxf(mx, pp[8 + i] * INVSH);
#pragma unroll
    for (int i = 0; i < 8; ++i) { e[i] = expf(pp[8 + i] * INVSH - mx); s += e[i]; }
    cum = 0.f; invs = 1.f / s;
    ch[0] = -TAILC;
#pragma unroll
    for (int i = 0; i < 8; ++i) {
      cum += 0.001f + 0.992f * e[i] * invs;
      ch[i + 1] = -TAILC + 2.f * TAILC * cum;
    }
    ch[8] = TAILC;
    // derivatives: pad value gives MIN_D + softplus(log(e^{0.999}-1)) == 1 exactly
    der[0] = 1.f; der[8] = 1.f;
#pragma unroll
    for (int i = 0; i < 7; ++i) {
      const float u = pp[16 + i];
      const float sp = (u > 15.f) ? u : log1pf(expf(u));
      der[i + 1] = 0.001f + sp;
    }

    const float zin = fminf(fmaxf(zv, -TAILC), TAILC);
    int idx = 0;
#pragma unroll
    for (int k = 1; k < 8; ++k) idx += (zin >= cw[k]) ? 1 : 0;
    idx += (zin >= (TAILC + 1e-6f)) ? 1 : 0;
    if (idx > 7) idx = 7;

    const float in_cw = cw[idx], in_w = cw[idx + 1] - cw[idx];
    const float in_ch = ch[idx], in_h = ch[idx + 1] - ch[idx];
    const float dl = in_h / in_w;
    const float d0 = der[idx], d1 = der[idx + 1];
    const float th  = (zin - in_cw) / in_w;
    const float th1 = th * (1.f - th);
    const float num = in_h * (dl * th * th + d0 * th1);
    const float den = dl + (d0 + d1 - 2.f * dl) * th1;
    float outv = in_ch + num / den;
    const float omth = 1.f - th;
    const float dnum = dl * dl * (d1 * th * th + 2.f * dl * th1 + d0 * omth * omth);
    float ladv = logf(dnum) - 2.f * logf(den);
    const bool inside = (zv >= -TAILC) && (zv <= TAILC);
    outv = inside ? outv : zv;
    ladv = inside ? ladv : 0.f;

    cur[(size_t)(rowBase + r) * ZD + d] = outv;
    atomicAdd(&sLad[r], ladv);
  }
  __syncthreads();
  if (tid < BM) lad[rowBase + tid] += sLad[tid];
}

// ---------------- finalize: log base density + accumulated log|det| ----------------
__global__ void finalize_kernel(const float* __restrict__ cur, const float* __restrict__ lad,
                                float* __restrict__ out) {
  const int b = blockIdx.x * blockDim.x + threadIdx.x;
  if (b >= B_SZ) return;
  float s = 0.f;
  const float* c = cur + (size_t)b * ZD;
#pragma unroll
  for (int d = 0; d < ZD; ++d) s += c[d] * c[d];
  out[b] = lad[b] - 0.5f * s - 58.81206612509905f;   // 0.5*64*ln(2*pi)
}

// ---------------- host launcher ----------------
extern "C" void kernel_launch(void* const* d_in, const int* in_sizes, int n_in,
                              void* d_out, int out_size, void* d_ws, size_t ws_size,
                              hipStream_t stream) {
  const float* z  = (const float*)d_in[0];
  const float* x  = (const float*)d_in[1];
  const float* W1 = (const float*)d_in[2];
  const float* b1 = (const float*)d_in[3];
  const float* Wc = (const float*)d_in[4];
  const float* bc = (const float*)d_in[5];
  const float* W2 = (const float*)d_in[6];
  const float* b2 = (const float*)d_in[7];
  const float* W3 = (const float*)d_in[8];
  const float* b3 = (const float*)d_in[9];
  float* out = (float*)d_out;

  char* ws = (char*)d_ws;
  unsigned short* xbf = (unsigned short*)ws; ws += (size_t)B_SZ * XD * 2;
  float* cur = (float*)ws;                   ws += (size_t)B_SZ * ZD * 4;
  float* lad = (float*)ws;                   ws += (size_t)B_SZ * 4;
  unsigned short* w1m = (unsigned short*)ws; ws += (size_t)TL * HID * ZD * 2;
  unsigned short* wcm = (unsigned short*)ws; ws += (size_t)TL * HID * XD * 2;
  unsigned short* w2m = (unsigned short*)ws; ws += (size_t)TL * HID * HID * 2;
  unsigned short* w3m = (unsigned short*)ws; ws += (size_t)TL * NP * HID * 2;

  const int thr = 256;
  prep_w1<<<(TL * HID * ZD + thr - 1) / thr, thr, 0, stream>>>(W1, w1m);
  prep_wc<<<(TL * HID * XD + thr - 1) / thr, thr, 0, stream>>>(Wc, wcm);
  prep_w2<<<(TL * HID * HID + thr - 1) / thr, thr, 0, stream>>>(W2, w2m);
  prep_w3<<<(TL * NP * HID + thr - 1) / thr, thr, 0, stream>>>(W3, w3m);
  prep_x <<<(B_SZ * XD + thr - 1) / thr, thr, 0, stream>>>(x, xbf);
  prep_init<<<(B_SZ * ZD + thr - 1) / thr, thr, 0, stream>>>(z, cur, lad);

  const size_t smem = (size_t)BM * KA * 2 + (size_t)BM * HID * 2 * 2 +
                      (size_t)BM * NP * 4 + (size_t)BM * ZD * 4 + (size_t)BM * 4;
  for (int t = 0; t < TL; ++t) {
    layer_kernel<<<B_SZ / BM, THREADS, smem, stream>>>(
        xbf, cur, lad,
        w1m + (size_t)t * HID * ZD,
        wcm + (size_t)t * HID * XD,
        w2m + (size_t)t * HID * HID,
        w3m + (size_t)t * NP * HID,
        b1 + t * HID, bc + t * HID, b2 + t * HID, b3 + t * NP);
  }
  finalize_kernel<<<(B_SZ + thr - 1) / thr, thr, 0, stream>>>(cur, lad, out);
}
